// NTXentLoss_24618752541169
// MI455X (gfx1250) — compile-verified, matrix operand order checked by reading, not run on hardware
//
#include <hip/hip_runtime.h>
#include <hip/hip_bf16.h>
#include <math.h>

#define NROWS   4096
#define DIM     256
#define TWO_N   8192
#define NTILES  (TWO_N / 16)      // 512 16x16 tiles per side
#define INV_T   (1.0f / 0.07f)
// fold 1/T and log2(e) into one constant: exp(x/T) == exp2(x * SCALE)
#define SCALE_LOG2 (INV_T * 1.4426950408889634f)

typedef __bf16 bf16_t;
typedef bf16_t bf16x8 __attribute__((ext_vector_type(8)));
typedef bf16_t v16bf  __attribute__((ext_vector_type(16)));
typedef float  v8f    __attribute__((ext_vector_type(8)));

// Raw v_exp_f32: safe here because |x| <= ~21 << 126 (no denormal-range guard
// needed, unlike libm exp2f which emits cmp/cndmask rescue chains).
__device__ __forceinline__ float fast_exp2(float x) {
    return __builtin_amdgcn_exp2f(x);
}

// ---------------------------------------------------------------------------
// Kernel 1: row L2-normalize z = [z1; z2], write bf16 copy for WMMA.
// One wave32 per row; each lane owns 8 contiguous elements.
// ---------------------------------------------------------------------------
__global__ __launch_bounds__(256) void ntx_normalize(const float* __restrict__ z1,
                                                     const float* __restrict__ z2,
                                                     bf16_t* __restrict__ zn) {
    int wave = (blockIdx.x * blockDim.x + threadIdx.x) >> 5;   // row id
    int lane = threadIdx.x & 31;
    if (wave >= TWO_N) return;
    const float* src = (wave < NROWS) ? (z1 + (size_t)wave * DIM)
                                      : (z2 + (size_t)(wave - NROWS) * DIM);
    float v[8];
    float ss = 0.0f;
#pragma unroll
    for (int j = 0; j < 8; ++j) {
        v[j] = src[lane * 8 + j];
        ss += v[j] * v[j];
    }
#pragma unroll
    for (int m = 1; m < 32; m <<= 1) ss += __shfl_xor(ss, m, 32);
    float inv = rsqrtf(ss);
    bf16x8 o;
#pragma unroll
    for (int j = 0; j < 8; ++j) o[j] = (bf16_t)(v[j] * inv);
    *(bf16x8*)(zn + (size_t)wave * DIM + lane * 8) = o;
}

// ---------------------------------------------------------------------------
// Per-lane fragment load for v_wmma_f32_16x16x32_bf16.
// A (16x32) and B (32x16, column-major view of z rows) use the SAME pattern:
//   lane (m = L&15, half = L>>4):
//     VGPR0..3  = row[k0 + half*8      .. +7]   (one b128 load)
//     VGPR4..7  = row[k0 + 16 + half*8 .. +7]   (one b128 load)
// ---------------------------------------------------------------------------
__device__ __forceinline__ v16bf load_frag(const bf16_t* __restrict__ row,
                                           int k0, int half) {
    bf16x8 lo = *(const bf16x8*)(row + k0 + half * 8);
    bf16x8 hi = *(const bf16x8*)(row + k0 + 16 + half * 8);
    return __builtin_shufflevector(lo, hi, 0, 1, 2, 3, 4, 5, 6, 7,
                                           8, 9, 10, 11, 12, 13, 14, 15);
}

// ---------------------------------------------------------------------------
// Kernel 2: fused Gram-matrix + softmax denominator.
// One wave per 16-row strip (512 strips). A strip's 8 K-fragments stay
// resident (64 VGPRs). Hot loop over 512 column tiles is fully UNMASKED:
// 16 b128 loads + 8 WMMAs + 4 pk_mul + 8 v_exp_f32 + 8 adds per tile.
// The diagonal tile's contribution is removed afterwards by recomputing it
// from the resident A fragments (B == A when cb == rb): 8 extra WMMAs,
// zero loads, no per-iteration compare/select chains in the hot loop.
// C layout: lane L holds column N=L&15, rows M = r + 8*(L>>4), r=0..7.
// ---------------------------------------------------------------------------
__global__ __launch_bounds__(256) void ntx_simsum(const bf16_t* __restrict__ zn,
                                                  float* __restrict__ rowsum) {
    int rb   = blockIdx.x * 8 + (threadIdx.x >> 5);   // row-tile index 0..511
    int lane = threadIdx.x & 31;
    int m    = lane & 15;
    int half = lane >> 4;

    const bf16_t* arow = zn + (size_t)(rb * 16 + m) * DIM;
    v16bf A[8];
#pragma unroll
    for (int kk = 0; kk < 8; ++kk) A[kk] = load_frag(arow, kk * 32, half);

    float acc[8];
#pragma unroll
    for (int r = 0; r < 8; ++r) acc[r] = 0.0f;

    // ---- hot loop: no diagonal logic, no guarded exp ----
    for (int cb = 0; cb < NTILES; ++cb) {
        const bf16_t* brow = zn + (size_t)(cb * 16 + m) * DIM;
        v8f c = {};
#pragma unroll
        for (int kk = 0; kk < 8; ++kk) {
            v16bf B = load_frag(brow, kk * 32, half);
            c = __builtin_amdgcn_wmma_f32_16x16x32_bf16(
                    false, A[kk], false, B, (short)0, c, false, false);
        }
#pragma unroll
        for (int r = 0; r < 8; ++r)
            acc[r] += fast_exp2(c[r] * SCALE_LOG2);   // 1 mul + raw v_exp_f32
    }

    // ---- diagonal correction: recompute tile cb == rb (B frag == A frag),
    //      bit-identical to the loop's version, and subtract its diag exp ----
    {
        v8f cd = {};
#pragma unroll
        for (int kk = 0; kk < 8; ++kk)
            cd = __builtin_amdgcn_wmma_f32_16x16x32_bf16(
                    false, A[kk], false, A[kk], (short)0, cd, false, false);
#pragma unroll
        for (int r = 0; r < 8; ++r) {
            // lane's element is sim[rb*16 + r + 8*half, rb*16 + m]
            float e = fast_exp2(cd[r] * SCALE_LOG2);
            acc[r] -= ((r + 8 * half) == m) ? e : 0.0f;
        }
    }

    // reduce across the 16 column-lanes of each half (xor<16 stays in-half)
#pragma unroll
    for (int r = 0; r < 8; ++r) {
        float a = acc[r];
#pragma unroll
        for (int mm = 1; mm < 16; mm <<= 1) a += __shfl_xor(a, mm, 32);
        if (m == 0) rowsum[rb * 16 + r + 8 * half] = a;
    }
}

// ---------------------------------------------------------------------------
// Kernel 3: positives  pos[i] = pos[i+N] = <zn[i], zn[i+N]> / T
// ---------------------------------------------------------------------------
__global__ __launch_bounds__(256) void ntx_pos(const bf16_t* __restrict__ zn,
                                               float* __restrict__ pos) {
    int wave = (blockIdx.x * blockDim.x + threadIdx.x) >> 5;
    int lane = threadIdx.x & 31;
    if (wave >= NROWS) return;
    const bf16_t* a = zn + (size_t)wave * DIM;
    const bf16_t* b = zn + (size_t)(wave + NROWS) * DIM;
    float s = 0.0f;
#pragma unroll
    for (int j = 0; j < 8; ++j)
        s += (float)a[lane * 8 + j] * (float)b[lane * 8 + j];
#pragma unroll
    for (int mm = 1; mm < 32; mm <<= 1) s += __shfl_xor(s, mm, 32);
    if (lane == 0) {
        float p = s * INV_T;
        pos[wave]          = p;
        pos[wave + NROWS]  = p;
    }
}

// ---------------------------------------------------------------------------
// Kernel 4: loss = mean_i ( log(denom_i) - pos_i )
// ---------------------------------------------------------------------------
__global__ __launch_bounds__(256) void ntx_reduce(const float* __restrict__ rowsum,
                                                  const float* __restrict__ pos,
                                                  float* __restrict__ out) {
    __shared__ float sm[256];
    float s = 0.0f;
    for (int i = threadIdx.x; i < TWO_N; i += 256)
        s += logf(rowsum[i]) - pos[i];
    sm[threadIdx.x] = s;
    __syncthreads();
    for (int off = 128; off > 0; off >>= 1) {
        if (threadIdx.x < off) sm[threadIdx.x] += sm[threadIdx.x + off];
        __syncthreads();
    }
    if (threadIdx.x == 0) out[0] = sm[0] / (float)TWO_N;
}

extern "C" void kernel_launch(void* const* d_in, const int* in_sizes, int n_in,
                              void* d_out, int out_size, void* d_ws, size_t ws_size,
                              hipStream_t stream) {
    const float* z1 = (const float*)d_in[0];
    const float* z2 = (const float*)d_in[1];
    float* out = (float*)d_out;

    char* ws = (char*)d_ws;
    bf16_t* zn     = (bf16_t*)ws;                              // 8192*256*2 = 4 MiB
    float*  rowsum = (float*)(ws + (size_t)TWO_N * DIM * 2);   // 32 KiB
    float*  pos    = rowsum + TWO_N;                           // 32 KiB

    ntx_normalize<<<TWO_N / 8, 256, 0, stream>>>(z1, z2, zn);
    ntx_simsum  <<<NTILES / 8, 256, 0, stream>>>(zn, rowsum);
    ntx_pos     <<<NROWS / 8,  256, 0, stream>>>(zn, pos);
    ntx_reduce  <<<1,          256, 0, stream>>>(rowsum, pos, out);
}
